// Attention_90305982366222
// MI455X (gfx1250) — compile-verified, matrix operand order checked by reading, not run on hardware
//
#include <hip/hip_runtime.h>
#include <hip/hip_bf16.h>
#include <stdint.h>

// ---------------------------------------------------------------------------
// Types for CDNA5 WMMA (wave32): v_wmma_f32_16x16x32_bf16
// ---------------------------------------------------------------------------
typedef __bf16 bf16_t;
typedef __attribute__((ext_vector_type(16))) __bf16 v16bf;
typedef __attribute__((ext_vector_type(8)))  __bf16 v8bf;
typedef __attribute__((ext_vector_type(8)))  float  v8f;
typedef __attribute__((ext_vector_type(4)))  float  v4f;
typedef __attribute__((ext_vector_type(4)))  unsigned int v4u;
typedef __attribute__((ext_vector_type(8)))  int v8i;
typedef __attribute__((ext_vector_type(4)))  int v4i;

#define WMMA_BF16(a, b, c) \
  __builtin_amdgcn_wmma_f32_16x16x32_bf16(false, (a), false, (b), (short)0, (c), false, false)

// Problem constants (fixed by the reference)
constexpr int Bc    = 8;
constexpr int Nc    = 4096;
constexpr int Cc    = 320;
constexpr int HEADc = 5;
constexpr int Dc    = 64;      // head dim
constexpr int NKVc  = 1024;    // N / SR^2
constexpr int KPATCH = 1280;   // 2*2*320
constexpr float SCALEc = 0.125f;   // 64^-0.5
constexpr float LNEPS  = 1e-5f;

// ---------------------------------------------------------------------------
// Load one 16x32 bf16 tile in WMMA A/B per-lane layout from row-major memory
// (global or LDS).  Lane L: row = rowBase + (L&15), g = L>>4; elements
// e<8 -> K = k0+8g+e, e>=8 -> K = k0+16+8g+(e-8).  Two 16-byte loads/lane.
// ---------------------------------------------------------------------------
__device__ inline v16bf load_tile16(const bf16_t* __restrict__ base, int ld,
                                    int rowBase, int k0) {
  const int lane = threadIdx.x & 31;
  const int g = lane >> 4;
  const bf16_t* p = base + (size_t)(rowBase + (lane & 15)) * ld + (k0 + g * 8);
  union { v16bf v; v8bf h[2]; } u;
  u.h[0] = *reinterpret_cast<const v8bf*>(p);
  u.h[1] = *reinterpret_cast<const v8bf*>(p + 16);
  return u.v;
}

// ---------------------------------------------------------------------------
// TDM: issue a 2-D tensor tile load (bf16, data_size=2B) into LDS.
// D# per CDNA5 ISA ch.8: group0 = {flags, lds_addr, global_addr, type=2},
// group1 = {data_size, tensor dims, tile dims, strides}; groups 2/3 zero (2D).
// Toolchain here uses the 6-arg builtin: (v4u, v8i, v4i, v4i, v8i, i32 cpol).
// ---------------------------------------------------------------------------
__device__ inline void tdm_load_2d_bf16(uint32_t ldsOff, const void* gptr,
                                        uint32_t tensorD0, uint32_t tensorD1,
                                        uint32_t tileD0, uint32_t tileD1,
                                        uint32_t stride0) {
  const uint64_t ga = (uint64_t)(uintptr_t)gptr;
  v4u g0;
  g0[0] = 1u;                                           // count=1 (valid), user desc
  g0[1] = ldsOff;                                       // lds_addr (bytes)
  g0[2] = (uint32_t)ga;                                 // global_addr[31:0]
  g0[3] = (uint32_t)((ga >> 32) & 0x1FFFFFFu) | (2u << 30);  // addr[56:32], type=2
  v8i g1;
  g1[0] = (int)(1u << 16);                              // data_size = 1 -> 2 bytes
  g1[1] = (int)((tensorD0 & 0xFFFFu) << 16);            // tensor_dim0[15:0] @48
  g1[2] = (int)(((tensorD0 >> 16) & 0xFFFFu) |          // tensor_dim0[31:16]
                ((tensorD1 & 0xFFFFu) << 16));          // tensor_dim1[15:0]
  g1[3] = (int)(((tensorD1 >> 16) & 0xFFFFu) |          // tensor_dim1[31:16]
                ((tileD0 & 0xFFFFu) << 16));            // tile_dim0
  g1[4] = (int)(tileD1 & 0xFFFFu);                      // tile_dim1 (tile_dim2=0)
  g1[5] = (int)stride0;                                 // tensor_dim0_stride[31:0]
  g1[6] = 0;                                            // stride0 hi / stride1 lo
  g1[7] = 0;                                            // stride1 hi
  const v4i z4 = {0, 0, 0, 0};
  const v8i z8 = {0, 0, 0, 0, 0, 0, 0, 0};
  __builtin_amdgcn_tensor_load_to_lds(g0, g1, z4, z4, z8, 0);
}

// ---------------------------------------------------------------------------
// Prep: x (f32) -> x_bf16 row-major, and 2x2-patch matrix for the strided conv
// ---------------------------------------------------------------------------
__global__ __launch_bounds__(256) void prep_x_kernel(
    const float* __restrict__ x, bf16_t* __restrict__ xbf,
    bf16_t* __restrict__ patches, int total) {
  int idx = blockIdx.x * blockDim.x + threadIdx.x;
  if (idx >= total) return;
  bf16_t v = (bf16_t)x[idx];
  xbf[idx] = v;
  int c = idx % Cc;
  int n = (idx / Cc) % Nc;
  int b = idx / (Cc * Nc);
  int y = n >> 6, xc = n & 63;            // H=W=64
  int i = y >> 1, h = y & 1, j = xc >> 1, w = xc & 1;
  size_t prow = (size_t)b * 1024 + (size_t)i * 32 + j;
  patches[prow * KPATCH + (size_t)((h * 2 + w) * Cc + c)] = v;
}

// ---------------------------------------------------------------------------
// Weight transpose + convert: in f32 [K][N] row-major -> out bf16 [N][K]
// ---------------------------------------------------------------------------
__global__ __launch_bounds__(256) void transpose_w_kernel(
    const float* __restrict__ in, bf16_t* __restrict__ out, int K, int N) {
  int idx = blockIdx.x * blockDim.x + threadIdx.x;
  if (idx >= N * K) return;
  int n = idx / K, k = idx % K;
  out[idx] = (bf16_t)in[(size_t)k * N + n];
}

// ---------------------------------------------------------------------------
// Generic WMMA GEMM:  D = A[M,K] * Bt[N,K]^T + bias[N]
// 8 waves/block; each wave computes a 32(M) x 64(N) tile (8 accumulators,
// B-tiles reused across two A-tiles); block tile 256 x 64.
// Epilogue modes:
//   0: f32 row-major -> outF
//   2: q scatter  -> outB0 bf16 [b][head][n][d]
//   3: kv scatter -> k: outB0 bf16 [b][head][m][d], v: outB1 bf16 [b][head][d][m]
// ---------------------------------------------------------------------------
__global__ __launch_bounds__(256) void gemm_bf16_kernel(
    const bf16_t* __restrict__ A, const bf16_t* __restrict__ Bt,
    const float* __restrict__ bias, int M, int N, int K, int mode,
    float* __restrict__ outF, bf16_t* __restrict__ outB0,
    bf16_t* __restrict__ outB1) {
  const int wave = threadIdx.x >> 5;
  const int lane = threadIdx.x & 31;
  const int mBase = blockIdx.x * 256 + wave * 32;
  const int nBase = blockIdx.y * 64;
  if (mBase >= M) return;  // wave-uniform

  v8f acc[2][4] = {};
  for (int k0 = 0; k0 < K; k0 += 32) {
    v16bf a0 = load_tile16(A, K, mBase, k0);
    v16bf a1 = load_tile16(A, K, mBase + 16, k0);
    v16bf bt[4];
#pragma unroll
    for (int j = 0; j < 4; ++j) bt[j] = load_tile16(Bt, K, nBase + 16 * j, k0);
#pragma unroll
    for (int j = 0; j < 4; ++j) {
      acc[0][j] = WMMA_BF16(a0, bt[j], acc[0][j]);
      acc[1][j] = WMMA_BF16(a1, bt[j], acc[1][j]);
    }
  }

  union AccU { v8f v; float f[8]; };
  const int half = lane >> 4, cb = lane & 15;
#pragma unroll
  for (int i = 0; i < 2; ++i) {
#pragma unroll
    for (int j = 0; j < 4; ++j) {
      AccU u; u.v = acc[i][j];
      const int col = nBase + j * 16 + cb;
      const float bv = bias[col];
#pragma unroll
      for (int r = 0; r < 8; ++r) {
        const int row = mBase + 16 * i + r + 8 * half;
        const float val = u.f[r] + bv;
        if (mode == 0) {
          outF[(size_t)row * N + col] = val;
        } else if (mode == 2) {
          int b = row >> 12, n = row & 4095, h = col >> 6, d = col & 63;
          outB0[(((size_t)b * HEADc + h) * Nc + n) * Dc + d] = (bf16_t)val;
        } else {  // mode 3: kv de-interleave, v stored transposed
          int b = row >> 10, m = row & 1023;
          if (col < Cc) {
            int h = col >> 6, d = col & 63;
            outB0[(((size_t)b * HEADc + h) * NKVc + m) * Dc + d] = (bf16_t)val;
          } else {
            int c2 = col - Cc, h = c2 >> 6, d = c2 & 63;
            outB1[(((size_t)b * HEADc + h) * Dc + d) * NKVc + m] = (bf16_t)val;
          }
        }
      }
    }
  }
}

// ---------------------------------------------------------------------------
// LayerNorm over C=320; one wave per row (10 elements/lane), bf16 output
// ---------------------------------------------------------------------------
__global__ __launch_bounds__(256) void ln_kernel(
    const float* __restrict__ xin, const float* __restrict__ gam,
    const float* __restrict__ bet, bf16_t* __restrict__ out, int rows) {
  const int lane = threadIdx.x & 31;
  const int row = blockIdx.x * 8 + (threadIdx.x >> 5);
  if (row >= rows) return;
  const float* p = xin + (size_t)row * Cc;
  float v[10];
  float s = 0.f;
#pragma unroll
  for (int i = 0; i < 10; ++i) { v[i] = p[i * 32 + lane]; s += v[i]; }
#pragma unroll
  for (int m = 16; m >= 1; m >>= 1) s += __shfl_xor(s, m);
  const float mu = s * (1.0f / Cc);
  float var = 0.f;
#pragma unroll
  for (int i = 0; i < 10; ++i) { float d = v[i] - mu; var += d * d; }
#pragma unroll
  for (int m = 16; m >= 1; m >>= 1) var += __shfl_xor(var, m);
  const float rs = rsqrtf(var * (1.0f / Cc) + LNEPS);
  bf16_t* o = out + (size_t)row * Cc;
#pragma unroll
  for (int i = 0; i < 10; ++i) {
    int col = i * 32 + lane;
    o[col] = (bf16_t)((v[i] - mu) * rs * gam[col] + bet[col]);
  }
}

// ---------------------------------------------------------------------------
// Flash attention with TDM-staged K/V:
// grid (Nq/128, B*HEAD); 8 waves/block, 16 q-rows per wave.
// Wave 0 issues tensor_load_to_lds for K (32x64) and V^T (64x32) chunks,
// double-buffered; all waves read the B-operands from LDS.  S never leaves
// the WGP (registers + 2KB/wave LDS score slab).
// ---------------------------------------------------------------------------
__global__ __launch_bounds__(256) void flash_kernel(
    const bf16_t* __restrict__ q, const bf16_t* __restrict__ k,
    const bf16_t* __restrict__ vT, bf16_t* __restrict__ outBNC) {
  __shared__ bf16_t kTile[2][32 * 64];   // [chunk row][d]
  __shared__ bf16_t vTile[2][64 * 32];   // [d][chunk col]
  __shared__ float  sS[8 * 512];         // per-wave 16x32 f32 score slab
  const int wave = threadIdx.x >> 5;
  const int lane = threadIdx.x & 31;
  const int half = lane >> 4, cb = lane & 15;
  const int bh = blockIdx.y;
  const int b = bh / HEADc, h = bh % HEADc;
  const int qBase = blockIdx.x * 128 + wave * 16;

  const bf16_t* qp = q  + (size_t)bh * Nc * Dc;
  const bf16_t* kp = k  + (size_t)bh * NKVc * Dc;
  const bf16_t* vp = vT + (size_t)bh * Dc * NKVc;
  float* lds = sS + wave * 512;

  const uint32_t kOff[2] = { (uint32_t)(uintptr_t)(void*)kTile[0],
                             (uint32_t)(uintptr_t)(void*)kTile[1] };
  const uint32_t vOff[2] = { (uint32_t)(uintptr_t)(void*)vTile[0],
                             (uint32_t)(uintptr_t)(void*)vTile[1] };

  const v16bf qa0 = load_tile16(qp, Dc, qBase, 0);
  const v16bf qa1 = load_tile16(qp, Dc, qBase, 32);

  // prologue: stage chunk 0
  if (wave == 0) {
    tdm_load_2d_bf16(kOff[0], kp, Dc, NKVc, Dc, 32, Dc);
    tdm_load_2d_bf16(vOff[0], vp, NKVc, Dc, 32, Dc, NKVc);
  }

  v8f o0 = {}, o1 = {}, o2 = {}, o3 = {};
  float mrow = -3.0e38f, lrow = 0.f;  // per-lane stats for q-row (lane&15)

  constexpr int nCh = NKVc / 32;
  for (int i = 0; i < nCh; ++i) {
    const int cur = i & 1;
    if (wave == 0) {
      if (i + 1 < nCh) {
        const int kv1 = (i + 1) * 32;
        tdm_load_2d_bf16(kOff[cur ^ 1], kp + (size_t)kv1 * Dc, Dc, NKVc, Dc, 32, Dc);
        tdm_load_2d_bf16(vOff[cur ^ 1], vp + kv1, NKVc, Dc, 32, Dc, NKVc);
        __builtin_amdgcn_s_wait_tensorcnt(2);  // 2 in-flight for i+1; chunk i done
      } else {
        __builtin_amdgcn_s_wait_tensorcnt(0);
      }
    }
    __syncthreads();  // chunk i resident for all waves
    const bf16_t* kb = kTile[cur];
    const bf16_t* vb = vTile[cur];

    // ---- S = q * K^T  (16 x 32 scores, two 16x16 D tiles, K-dim = 64) ----
    v8f s0 = {}, s1 = {};
    {
      v16bf kb0 = load_tile16(kb, Dc, 0,  0);
      v16bf kb1 = load_tile16(kb, Dc, 16, 0);
      s0 = WMMA_BF16(qa0, kb0, s0);
      s1 = WMMA_BF16(qa0, kb1, s1);
      kb0 = load_tile16(kb, Dc, 0,  32);
      kb1 = load_tile16(kb, Dc, 16, 32);
      s0 = WMMA_BF16(qa1, kb0, s0);
      s1 = WMMA_BF16(qa1, kb1, s1);
    }
    // ---- D-layout -> LDS (scaled) ----
    union AccU { v8f v; float f[8]; };
    AccU us0, us1; us0.v = s0; us1.v = s1;
#pragma unroll
    for (int r = 0; r < 8; ++r) {
      int srow = r + 8 * half;
      lds[srow * 32 + cb]      = us0.f[r] * SCALEc;
      lds[srow * 32 + 16 + cb] = us1.f[r] * SCALEc;
    }
    // ---- re-read in A-operand layout: lane holds row (lane&15) ----
    const float* rp = lds + cb * 32;
    union V4U { v4f v; float f[4]; };
    V4U t0, t1, t2, t3;
    t0.v = *reinterpret_cast<const v4f*>(rp + 8 * half);
    t1.v = *reinterpret_cast<const v4f*>(rp + 8 * half + 4);
    t2.v = *reinterpret_cast<const v4f*>(rp + 16 + 8 * half);
    t3.v = *reinterpret_cast<const v4f*>(rp + 16 + 8 * half + 4);
    float sv[16];
#pragma unroll
    for (int e = 0; e < 4; ++e) {
      sv[e] = t0.f[e]; sv[4 + e] = t1.f[e];
      sv[8 + e] = t2.f[e]; sv[12 + e] = t3.f[e];
    }
    // ---- online softmax ----
    float cmax = sv[0];
#pragma unroll
    for (int e = 1; e < 16; ++e) cmax = fmaxf(cmax, sv[e]);
    cmax = fmaxf(cmax, __shfl_xor(cmax, 16));
    const float mNew = fmaxf(mrow, cmax);
    const float alpha = __expf(mrow - mNew);
    union PU { v16bf v; bf16_t e[16]; };
    PU pu;
    float ssum = 0.f;
#pragma unroll
    for (int e = 0; e < 16; ++e) {
      float pe = __expf(sv[e] - mNew);
      ssum += pe;
      pu.e[e] = (bf16_t)pe;
    }
    ssum += __shfl_xor(ssum, 16);
    lrow = lrow * alpha + ssum;
    mrow = mNew;
    // ---- rescale O by per-row alpha (broadcast row stat into D-layout) ----
    union AccU uo0, uo1, uo2, uo3;
    uo0.v = o0; uo1.v = o1; uo2.v = o2; uo3.v = o3;
#pragma unroll
    for (int r = 0; r < 8; ++r) {
      float ar = __shfl(alpha, r + 8 * half);
      uo0.f[r] *= ar; uo1.f[r] *= ar; uo2.f[r] *= ar; uo3.f[r] *= ar;
    }
    o0 = uo0.v; o1 = uo1.v; o2 = uo2.v; o3 = uo3.v;
    // ---- O += P * V  (V tile in LDS, [d][32]) ----
    o0 = WMMA_BF16(pu.v, load_tile16(vb, 32, 0,  0), o0);
    o1 = WMMA_BF16(pu.v, load_tile16(vb, 32, 16, 0), o1);
    o2 = WMMA_BF16(pu.v, load_tile16(vb, 32, 32, 0), o2);
    o3 = WMMA_BF16(pu.v, load_tile16(vb, 32, 48, 0), o3);
    __syncthreads();  // all waves done with buffer `cur`
  }

  // ---- normalize and store into [B][N][C] bf16 at column h*64 ----
  const float inv = 1.0f / lrow;
  union AccU2 { v8f v; float f[8]; };
  AccU2 uo[4];
  uo[0].v = o0; uo[1].v = o1; uo[2].v = o2; uo[3].v = o3;
#pragma unroll
  for (int r = 0; r < 8; ++r) {
    const float ir = __shfl(inv, r + 8 * half);
    const size_t row = (size_t)b * Nc + (size_t)(qBase + r + 8 * half);
#pragma unroll
    for (int j = 0; j < 4; ++j) {
      int col = h * 64 + j * 16 + cb;
      outBNC[row * Cc + col] = (bf16_t)(uo[j].f[r] * ir);
    }
  }
}

// ---------------------------------------------------------------------------
// Host launch
// ---------------------------------------------------------------------------
extern "C" void kernel_launch(void* const* d_in, const int* in_sizes, int n_in,
                              void* d_out, int out_size, void* d_ws,
                              size_t ws_size, hipStream_t stream) {
  (void)in_sizes; (void)n_in; (void)out_size; (void)ws_size;
  const float* x      = (const float*)d_in[0];
  const float* q_w    = (const float*)d_in[1];
  const float* q_b    = (const float*)d_in[2];
  const float* kv_w   = (const float*)d_in[3];
  const float* kv_b   = (const float*)d_in[4];
  const float* sr_w   = (const float*)d_in[5];
  const float* sr_b   = (const float*)d_in[6];
  const float* ln_g   = (const float*)d_in[7];
  const float* ln_b   = (const float*)d_in[8];
  const float* proj_w = (const float*)d_in[9];
  const float* proj_b = (const float*)d_in[10];

  char* ws = (char*)d_ws;
  size_t off = 0;
  auto alloc = [&](size_t bytes) -> void* {
    off = (off + 255) & ~(size_t)255;
    void* p = ws + off;
    off += bytes;
    return p;
  };

  const size_t BN = (size_t)Bc * Nc;          // 32768
  const size_t BM = (size_t)Bc * NKVc;        // 8192
  bf16_t* xbf    = (bf16_t*)alloc(BN * Cc * 2);
  bf16_t* patch  = (bf16_t*)alloc(BM * KPATCH * 2);
  bf16_t* qwT    = (bf16_t*)alloc((size_t)Cc * Cc * 2);
  bf16_t* kvwT   = (bf16_t*)alloc((size_t)2 * Cc * Cc * 2);
  bf16_t* srwT   = (bf16_t*)alloc((size_t)Cc * KPATCH * 2);
  bf16_t* pjwT   = (bf16_t*)alloc((size_t)Cc * Cc * 2);
  bf16_t* qbf    = (bf16_t*)alloc((size_t)Bc * HEADc * Nc * Dc * 2);
  float*  convf  = (float*) alloc(BM * Cc * 4);
  bf16_t* xln    = (bf16_t*)alloc(BM * Cc * 2);
  bf16_t* kbf    = (bf16_t*)alloc((size_t)Bc * HEADc * NKVc * Dc * 2);
  bf16_t* vTbf   = (bf16_t*)alloc((size_t)Bc * HEADc * Dc * NKVc * 2);
  bf16_t* attnbf = (bf16_t*)alloc(BN * Cc * 2);

  // 1. prep x + patches, transpose weights to [N][K] bf16
  {
    int total = (int)(BN * Cc);
    prep_x_kernel<<<(total + 255) / 256, 256, 0, stream>>>(x, xbf, patch, total);
    transpose_w_kernel<<<(Cc * Cc + 255) / 256, 256, 0, stream>>>(q_w, qwT, Cc, Cc);
    transpose_w_kernel<<<(Cc * 2 * Cc + 255) / 256, 256, 0, stream>>>(kv_w, kvwT, Cc, 2 * Cc);
    transpose_w_kernel<<<(KPATCH * Cc + 255) / 256, 256, 0, stream>>>(sr_w, srwT, KPATCH, Cc);
    transpose_w_kernel<<<(Cc * Cc + 255) / 256, 256, 0, stream>>>(proj_w, pjwT, Cc, Cc);
  }
  // 2. q projection -> [b][head][n][d] bf16
  gemm_bf16_kernel<<<dim3((int)BN / 256, Cc / 64), 256, 0, stream>>>(
      xbf, qwT, q_b, (int)BN, Cc, Cc, 2, nullptr, qbf, nullptr);
  // 3. spatial-reduction conv as GEMM -> f32
  gemm_bf16_kernel<<<dim3((int)BM / 256, Cc / 64), 256, 0, stream>>>(
      patch, srwT, sr_b, (int)BM, Cc, KPATCH, 0, convf, nullptr, nullptr);
  // 4. LayerNorm -> bf16
  ln_kernel<<<(int)BM / 8, 256, 0, stream>>>(convf, ln_g, ln_b, xln, (int)BM);
  // 5. kv projection -> k [bh][m][d], v transposed [bh][d][m]
  gemm_bf16_kernel<<<dim3((int)BM / 256, 2 * Cc / 64), 256, 0, stream>>>(
      xln, kvwT, kv_b, (int)BM, 2 * Cc, Cc, 3, nullptr, kbf, vTbf);
  // 6. flash attention (TDM-staged K/V) -> [B][N][C] bf16
  flash_kernel<<<dim3(Nc / 128, Bc * HEADc), 256, 0, stream>>>(qbf, kbf, vTbf, attnbf);
  // 7. output projection -> f32 d_out
  gemm_bf16_kernel<<<dim3((int)BN / 256, Cc / 64), 256, 0, stream>>>(
      attnbf, pjwT, proj_b, (int)BN, Cc, Cc, 0, (float*)d_out, nullptr, nullptr);
}
